// GAT_83820581749190
// MI455X (gfx1250) — compile-verified, matrix operand order checked by reading, not run on hardware
//
#include <hip/hip_runtime.h>

// ---------------------------------------------------------------------------
// Problem constants (match reference)
// ---------------------------------------------------------------------------
#define N_NODES 50000
#define N_EDGES 800000
#define OUT_CH  10
#define N_GRAPH 64
#define NEG_SLOPE 0.2f
#define NEGBIG  (-3.0e38f)

typedef __attribute__((ext_vector_type(16))) __bf16 v16bf;
typedef __attribute__((ext_vector_type(8)))  float  v8f;

__device__ __forceinline__ float lrelu(float v) {
    return v > 0.0f ? v : NEG_SLOPE * v;
}
__device__ __forceinline__ float elu(float v) {
    return v > 0.0f ? v : (__expf(v) - 1.0f);
}

// ---------------------------------------------------------------------------
// Fills
// ---------------------------------------------------------------------------
__global__ void fill_f32(float* p, float v, int n) {
    int i = blockIdx.x * blockDim.x + threadIdx.x;
    if (i < n) p[i] = v;
}
__global__ void fill_u32(unsigned* p, unsigned v, int n) {
    int i = blockIdx.x * blockDim.x + threadIdx.x;
    if (i < n) p[i] = v;
}

// ---------------------------------------------------------------------------
// WMMA GEMM: C[M,N] = A[M,K] * B[K,N]. A,B fp32 row-major, converted to bf16
// on the fly, fp32 accumulate. K is compile-time -> K-loop fully unrolls into
// K/32 back-to-back v_wmma_f32_16x16x32_bf16. One wave per 16x16 output tile.
// ---------------------------------------------------------------------------
template <int K>
__global__ void wmma_gemm_bf16(const float* __restrict__ A,
                               const float* __restrict__ B,
                               float* __restrict__ C,
                               int M, int N) {
    const int wavesPerBlock = blockDim.x >> 5;
    int wave   = blockIdx.x * wavesPerBlock + (threadIdx.x >> 5);
    int tilesN = N >> 4;
    int tiles  = (M >> 4) * tilesN;
    if (wave >= tiles) return;
    int ti = wave / tilesN;          // row tile
    int tj = wave % tilesN;          // col tile
    int lane = threadIdx.x & 31;
    int rc   = lane & 15;            // row index for A, col index for B/C
    int g    = lane >> 4;            // lane half-group

    v8f acc = {};
    const float* Arow = A + (size_t)(ti * 16 + rc) * K;
#pragma unroll
    for (int kt = 0; kt < K; kt += 32) {
        v16bf a, b;
#pragma unroll
        for (int i = 0; i < 8; ++i) {
            // 16-bit A-matrix 16x32 layout: VGPR i holds K pair at
            // k = 2i + 8g (i<4)  or  16 + 2(i-4) + 8g (i>=4)
            int koff = (i < 4) ? (2 * i + 8 * g) : (16 + 2 * (i - 4) + 8 * g);
            int k = kt + koff;
            a[2 * i]     = (__bf16)Arow[k];
            a[2 * i + 1] = (__bf16)Arow[k + 1];
            b[2 * i]     = (__bf16)B[(size_t)k * N + tj * 16 + rc];
            b[2 * i + 1] = (__bf16)B[(size_t)(k + 1) * N + tj * 16 + rc];
        }
        acc = __builtin_amdgcn_wmma_f32_16x16x32_bf16(
            false, a, false, b, (short)0, acc, false, false);
    }
#pragma unroll
    for (int r = 0; r < 8; ++r) {
        int m = ti * 16 + r + 8 * g;     // f32 C/D layout: VGPR r -> M = r + 8g
        C[(size_t)m * N + tj * 16 + rc] = acc[r];
    }
}

// ---------------------------------------------------------------------------
// Attention coefficients: als[n,h] = <h[n,h,:], a_src[h,:]>, same for ald.
// One thread per (node, head).
// ---------------------------------------------------------------------------
__global__ void attn_coef(const float* __restrict__ h,
                          const float* __restrict__ a_src,
                          const float* __restrict__ a_dst,
                          float* __restrict__ als, float* __restrict__ ald,
                          int n, int heads, int ch) {
    int i = blockIdx.x * blockDim.x + threadIdx.x;
    if (i >= n * heads) return;
    int node = i / heads, hd = i % heads;
    const float* row = h + (size_t)node * heads * ch + hd * ch;
    const float* as  = a_src + hd * ch;
    const float* ad  = a_dst + hd * ch;
    float s = 0.f, d = 0.f;
    for (int c = 0; c < ch; ++c) { s += row[c] * as[c]; d += row[c] * ad[c]; }
    als[i] = s;
    ald[i] = d;
}

// ---------------------------------------------------------------------------
// CSR build over destination nodes. Edge ids >= E0 are self loops (src=dst).
// ---------------------------------------------------------------------------
__global__ void deg_count(const int* __restrict__ ei, int E0, int Eall,
                          unsigned* __restrict__ deg) {
    int e = blockIdx.x * blockDim.x + threadIdx.x;
    if (e >= Eall) return;
    int d = (e < E0) ? ei[E0 + e] : (e - E0);
    atomicAdd(&deg[d], 1u);
}

// Single-workgroup exclusive scan over n counts -> rowptr[n+1]; cursor = copy.
__global__ void scan_rowptr(const unsigned* __restrict__ deg,
                            unsigned* __restrict__ rowptr,
                            unsigned* __restrict__ cursor, int n) {
    __shared__ unsigned tmp[1024];
    __shared__ unsigned carry;
    int t = threadIdx.x;
    if (t == 0) carry = 0;
    __syncthreads();
    for (int base = 0; base < n; base += 1024) {
        int i = base + t;
        unsigned v = (i < n) ? deg[i] : 0u;
        tmp[t] = v;
        __syncthreads();
        for (int off = 1; off < 1024; off <<= 1) {
            unsigned u = (t >= off) ? tmp[t - off] : 0u;
            __syncthreads();
            tmp[t] += u;
            __syncthreads();
        }
        unsigned c = carry;
        unsigned excl = tmp[t] - v + c;
        if (i < n) { rowptr[i] = excl; cursor[i] = excl; }
        __syncthreads();
        if (t == 1023) carry = c + tmp[1023];
        __syncthreads();
    }
    if (t == 0) rowptr[n] = carry;
}

__global__ void csr_fill(const int* __restrict__ ei, int E0, int Eall,
                         unsigned* __restrict__ cursor,
                         int* __restrict__ srcl) {
    int e = blockIdx.x * blockDim.x + threadIdx.x;
    if (e >= Eall) return;
    int s = (e < E0) ? ei[e]      : (e - E0);
    int d = (e < E0) ? ei[E0 + e] : (e - E0);
    unsigned idx = atomicAdd(&cursor[d], 1u);
    srcl[idx] = s;
}

// ---------------------------------------------------------------------------
// Fused GAT layer 1 (H=4, C=128): one wave per destination node.
// Pass A/B: lanes stride over in-edges, wave-allreduce max and sum-of-exp per
// head. Pass C: lanes own 4 contiguous channels (float4 gather of h[src]),
// serial loop over edges, register accumulate; bias+ELU fused into the store.
// No atomics anywhere.
// ---------------------------------------------------------------------------
__global__ void gat_agg4(const unsigned* __restrict__ rowptr,
                         const int* __restrict__ srcl,
                         const float* __restrict__ hfeat,
                         const float* __restrict__ als,
                         const float* __restrict__ ald,
                         const float* __restrict__ bias,
                         float* __restrict__ out, int n) {
    int wid  = (blockIdx.x * blockDim.x + threadIdx.x) >> 5;
    int lane = threadIdx.x & 31;
    if (wid >= n) return;
    unsigned r0 = rowptr[wid], r1 = rowptr[wid + 1];
    float ad0 = ald[wid * 4 + 0], ad1 = ald[wid * 4 + 1];
    float ad2 = ald[wid * 4 + 2], ad3 = ald[wid * 4 + 3];

    // pass A: per-head max over incoming edges
    float m0 = NEGBIG, m1 = NEGBIG, m2 = NEGBIG, m3 = NEGBIG;
    for (unsigned i = r0 + lane; i < r1; i += 32) {
        int s = srcl[i];
        m0 = fmaxf(m0, lrelu(als[s * 4 + 0] + ad0));
        m1 = fmaxf(m1, lrelu(als[s * 4 + 1] + ad1));
        m2 = fmaxf(m2, lrelu(als[s * 4 + 2] + ad2));
        m3 = fmaxf(m3, lrelu(als[s * 4 + 3] + ad3));
    }
#pragma unroll
    for (int off = 16; off; off >>= 1) {
        m0 = fmaxf(m0, __shfl_xor(m0, off, 32));
        m1 = fmaxf(m1, __shfl_xor(m1, off, 32));
        m2 = fmaxf(m2, __shfl_xor(m2, off, 32));
        m3 = fmaxf(m3, __shfl_xor(m3, off, 32));
    }
    // pass B: per-head sum of exp
    float s0 = 0.f, s1 = 0.f, s2 = 0.f, s3 = 0.f;
    for (unsigned i = r0 + lane; i < r1; i += 32) {
        int s = srcl[i];
        s0 += __expf(lrelu(als[s * 4 + 0] + ad0) - m0);
        s1 += __expf(lrelu(als[s * 4 + 1] + ad1) - m1);
        s2 += __expf(lrelu(als[s * 4 + 2] + ad2) - m2);
        s3 += __expf(lrelu(als[s * 4 + 3] + ad3) - m3);
    }
#pragma unroll
    for (int off = 16; off; off >>= 1) {
        s0 += __shfl_xor(s0, off, 32);
        s1 += __shfl_xor(s1, off, 32);
        s2 += __shfl_xor(s2, off, 32);
        s3 += __shfl_xor(s3, off, 32);
    }
    // pass C: aggregate. lane owns channels [c0, c0+4); head = c0/32.
    int c0 = lane << 2;
    int hs = c0 >> 5;
    float mh  = hs == 0 ? m0  : hs == 1 ? m1  : hs == 2 ? m2  : m3;
    float sh  = hs == 0 ? s0  : hs == 1 ? s1  : hs == 2 ? s2  : s3;
    float adh = hs == 0 ? ad0 : hs == 1 ? ad1 : hs == 2 ? ad2 : ad3;
    float inv = 1.0f / sh;
    float ax = 0.f, ay = 0.f, az = 0.f, aw = 0.f;
    for (unsigned i = r0; i < r1; ++i) {
        int s = srcl[i];
        float alpha = __expf(lrelu(als[s * 4 + hs] + adh) - mh) * inv;
        const float4 hv = *(const float4*)(hfeat + (size_t)s * 128 + c0);
        ax += hv.x * alpha;
        ay += hv.y * alpha;
        az += hv.z * alpha;
        aw += hv.w * alpha;
    }
    const float4 b4 = *(const float4*)(bias + c0);
    float4 o;
    o.x = elu(ax + b4.x);
    o.y = elu(ay + b4.y);
    o.z = elu(az + b4.z);
    o.w = elu(aw + b4.w);
    *(float4*)(out + (size_t)wid * 128 + c0) = o;
}

// Fused GAT layer 2 (H=1, C=32): one wave per destination, one channel/lane.
__global__ void gat_agg1(const unsigned* __restrict__ rowptr,
                         const int* __restrict__ srcl,
                         const float* __restrict__ hfeat,
                         const float* __restrict__ als,
                         const float* __restrict__ ald,
                         const float* __restrict__ bias,
                         float* __restrict__ out, int n) {
    int wid  = (blockIdx.x * blockDim.x + threadIdx.x) >> 5;
    int lane = threadIdx.x & 31;
    if (wid >= n) return;
    unsigned r0 = rowptr[wid], r1 = rowptr[wid + 1];
    float adn = ald[wid];

    float m = NEGBIG;
    for (unsigned i = r0 + lane; i < r1; i += 32)
        m = fmaxf(m, lrelu(als[srcl[i]] + adn));
#pragma unroll
    for (int off = 16; off; off >>= 1)
        m = fmaxf(m, __shfl_xor(m, off, 32));

    float sm = 0.f;
    for (unsigned i = r0 + lane; i < r1; i += 32)
        sm += __expf(lrelu(als[srcl[i]] + adn) - m);
#pragma unroll
    for (int off = 16; off; off >>= 1)
        sm += __shfl_xor(sm, off, 32);
    float inv = 1.0f / sm;

    float acc = 0.f;
    for (unsigned i = r0; i < r1; ++i) {
        int s = srcl[i];
        float alpha = __expf(lrelu(als[s] + adn) - m) * inv;
        acc += alpha * hfeat[(size_t)s * 32 + lane];
    }
    out[(size_t)wid * 32 + lane] = elu(acc + bias[lane]);
}

// global mean pool (accumulate): one thread per (node, channel)
__global__ void pool_accum(const float* __restrict__ h,
                           const int* __restrict__ batch,
                           float* __restrict__ gsum, float* __restrict__ gcnt,
                           int n) {
    int i = blockIdx.x * blockDim.x + threadIdx.x;
    if (i >= n * 32) return;
    int node = i >> 5, c = i & 31;
    int g = batch[node];
    atomicAdd(&gsum[g * 32 + c], h[i]);
    if (c == 0) atomicAdd(&gcnt[g], 1.0f);
}

// out[g,o] = (gsum[g,:]/max(cnt,1)) @ Wl + bl
__global__ void final_linear(const float* __restrict__ gsum,
                             const float* __restrict__ gcnt,
                             const float* __restrict__ Wl,
                             const float* __restrict__ bl,
                             float* __restrict__ out, int Gn, int C, int O) {
    int i = blockIdx.x * blockDim.x + threadIdx.x;
    if (i >= Gn * O) return;
    int g = i / O, o = i % O;
    float cnt = gcnt[g];
    cnt = cnt > 1.f ? cnt : 1.f;
    float acc = bl[o];
    for (int c = 0; c < C; ++c) acc += (gsum[g * C + c] / cnt) * Wl[c * O + o];
    out[i] = acc;
}

// ---------------------------------------------------------------------------
// Host launch
// ---------------------------------------------------------------------------
extern "C" void kernel_launch(void* const* d_in, const int* in_sizes, int n_in,
                              void* d_out, int out_size, void* d_ws, size_t ws_size,
                              hipStream_t stream) {
    const int N = N_NODES, E0 = N_EDGES, Eall = N_EDGES + N_NODES, Gn = N_GRAPH;

    const float* x   = (const float*)d_in[0];
    const int*   ei  = (const int*)d_in[1];
    const int*   bat = (const int*)d_in[2];
    const float* W1  = (const float*)d_in[3];
    const float* a1s = (const float*)d_in[4];
    const float* a1d = (const float*)d_in[5];
    const float* b1  = (const float*)d_in[6];
    const float* W2  = (const float*)d_in[7];
    const float* a2s = (const float*)d_in[8];
    const float* a2d = (const float*)d_in[9];
    const float* b2  = (const float*)d_in[10];
    const float* Wl  = (const float*)d_in[11];
    const float* bl  = (const float*)d_in[12];
    float* out = (float*)d_out;

    // workspace layout
    char* ws = (char*)d_ws;
    size_t off = 0;
    auto alloc = [&](size_t bytes) -> char* {
        char* p = ws + off;
        off += (bytes + 255) & ~(size_t)255;
        return p;
    };
    float*    h1     = (float*)alloc((size_t)N * 128 * 4);   // X@W1
    float*    als1   = (float*)alloc((size_t)N * 4 * 4);
    float*    ald1   = (float*)alloc((size_t)N * 4 * 4);
    float*    out1   = (float*)alloc((size_t)N * 128 * 4);   // layer-1 output (elu'd)
    float*    h2     = (float*)alloc((size_t)N * 32 * 4);    // out1@W2
    float*    als2   = (float*)alloc((size_t)N * 4);
    float*    ald2   = (float*)alloc((size_t)N * 4);
    float*    out2   = (float*)alloc((size_t)N * 32 * 4);
    unsigned* deg    = (unsigned*)alloc((size_t)N * 4);
    unsigned* rowptr = (unsigned*)alloc((size_t)(N + 1) * 4);
    unsigned* cursor = (unsigned*)alloc((size_t)N * 4);
    int*      srcl   = (int*)alloc((size_t)Eall * 4);
    float*    gsum   = (float*)alloc((size_t)Gn * 32 * 4);
    float*    gcnt   = (float*)alloc((size_t)Gn * 4);

    const int B = 256;
    auto blk = [](long long n) { return dim3((unsigned)((n + 255) / 256)); };

    // ---- re-init mutable scratch (deterministic across graph replays) ----
    fill_u32<<<blk(N), B, 0, stream>>>(deg, 0u, N);
    fill_f32<<<blk(Gn * 32), B, 0, stream>>>(gsum, 0.f, Gn * 32);
    fill_f32<<<blk(Gn), B, 0, stream>>>(gcnt, 0.f, Gn);

    // ---- CSR of destination adjacency (shared by both layers) ----
    deg_count<<<blk(Eall), B, 0, stream>>>(ei, E0, Eall, deg);
    scan_rowptr<<<1, 1024, 0, stream>>>(deg, rowptr, cursor, N);
    csr_fill<<<blk(Eall), B, 0, stream>>>(ei, E0, Eall, cursor, srcl);

    // ---- conv1: h1 = X @ W1 (WMMA bf16), attn coefs, fused softmax+agg ----
    {
        int tiles = (N / 16) * (128 / 16);                 // 25000 waves
        wmma_gemm_bf16<128><<<dim3((tiles + 7) / 8), B, 0, stream>>>(x, W1, h1, N, 128);
    }
    attn_coef<<<blk(N * 4), B, 0, stream>>>(h1, a1s, a1d, als1, ald1, N, 4, 32);
    gat_agg4<<<blk((long long)N * 32), B, 0, stream>>>(rowptr, srcl, h1, als1, ald1, b1, out1, N);

    // ---- conv2: h2 = out1 @ W2 (WMMA bf16), 1 head, fused softmax+agg ----
    {
        int tiles = (N / 16) * (32 / 16);                  // 6250 waves
        wmma_gemm_bf16<128><<<dim3((tiles + 7) / 8), B, 0, stream>>>(out1, W2, h2, N, 32);
    }
    attn_coef<<<blk(N), B, 0, stream>>>(h2, a2s, a2d, als2, ald2, N, 1, 32);
    gat_agg1<<<blk((long long)N * 32), B, 0, stream>>>(rowptr, srcl, h2, als2, ald2, b2, out2, N);

    // ---- global mean pool + final linear ----
    pool_accum<<<blk((long long)N * 32), B, 0, stream>>>(out2, bat, gsum, gcnt, N);
    final_linear<<<blk(Gn * OUT_CH), B, 0, stream>>>(gsum, gcnt, Wl, bl, out, Gn, 32, OUT_CH);
}
